// SPRGNN_88648124991074
// MI455X (gfx1250) — compile-verified, matrix operand order checked by reading, not run on hardware
//
#include <hip/hip_runtime.h>
#include <hip/hip_bf16.h>

#define N_NODES   100000
#define N_EDGES   1000000
#define N_GRAPHS  1024
#define NUM_CLASSES 10

typedef __attribute__((ext_vector_type(16))) __bf16 v16bf;
typedef __attribute__((ext_vector_type(8)))  __bf16 v8bf;
typedef __attribute__((ext_vector_type(4)))  __bf16 v4bf;
typedef __attribute__((ext_vector_type(8)))  float  v8f;
typedef __attribute__((ext_vector_type(4)))  unsigned int v4u;
typedef __attribute__((ext_vector_type(8)))  int v8i;
typedef __attribute__((ext_vector_type(4)))  int v4i;

union FragBF16 { v16bf v; v8bf h8[2]; };

// ---------------------------------------------------------------------------
// TDM: async 2D tile copy (32 rows x K bf16, row-major, contiguous rows) from
// global to LDS. Descriptor per ISA 8.3/8.4: group0 = {count/flags, lds_addr,
// global_addr lo/hi + type=2}, group1 = {data_size=2B, tensor_dim0=K,
// tensor_dim1=rows_remaining (OOB rows -> zero fill), tile 32xK, stride0=K}.
// Trailing groups zero (tile_dim2..4 unused). This toolchain exposes the
// 6-arg builtin: (v4u, v8i, v4i, v4i, v8i, i32 cpol).
// ---------------------------------------------------------------------------
template <int K>
__device__ inline void tdm_load_tile_bf16(const __bf16* gsrc, unsigned lds_off,
                                          unsigned rows_remaining) {
    unsigned long long ga = (unsigned long long)(uintptr_t)gsrc;
    v4u g0;
    g0[0] = 1u;                                              // count=1, user D#
    g0[1] = lds_off;                                         // lds_addr (bytes)
    g0[2] = (unsigned)(ga & 0xFFFFFFFFu);                    // global_addr[31:0]
    g0[3] = (unsigned)((ga >> 32) & 0x01FFFFFFu) | (2u << 30); // [56:32] | type=2
    v8i g1;
    g1[0] = (int)(1u << 16);                                 // data_size=1 (2B)
    g1[1] = (int)((unsigned)K << 16);                        // tensor_dim0[15:0]
    g1[2] = (int)((rows_remaining & 0xFFFFu) << 16);         // tdim0 hi=0 | tdim1 lo
    g1[3] = (int)(((rows_remaining >> 16) & 0xFFFFu) |
                  ((unsigned)K << 16));                      // tdim1 hi | tile_dim0=K
    g1[4] = 32;                                              // tile_dim1=32, tile_dim2=0
    g1[5] = K;                                               // tensor_dim0_stride lo
    g1[6] = 0;                                               // stride hi | dim1_stride lo
    g1[7] = 0;
    v4i z4 = {0, 0, 0, 0};
    v8i z8 = {0, 0, 0, 0, 0, 0, 0, 0};
    __builtin_amdgcn_tensor_load_to_lds(g0, g1, z4, z4, z8, 0);
}

// ---------------------------------------------------------------------------
// 0) zero-fill (agg / pooled re-zeroed every call: graph replay safe)
// ---------------------------------------------------------------------------
__global__ void fill_zero_kernel(float* __restrict__ p, long long n) {
    long long i = (long long)blockIdx.x * blockDim.x + threadIdx.x;
    if (i < n) p[i] = 0.0f;
}

// ---------------------------------------------------------------------------
// 1) embedding lookup + node MLP: h0 = relu([shape|color] @ W_node + b), bf16 out
// ---------------------------------------------------------------------------
__global__ void embed_mlp_kernel(const int*   __restrict__ x,
                                 const float* __restrict__ shape_emb,
                                 const float* __restrict__ color_emb,
                                 const float* __restrict__ W_node,   // (16,32)
                                 const float* __restrict__ b_node,   // (32,)
                                 __bf16*      __restrict__ h0)       // (N,32) bf16
{
    __shared__ float sW[16 * 32];
    __shared__ float sSh[8 * 8];
    __shared__ float sCo[8 * 8];
    __shared__ float sB[32];
    int tid = threadIdx.x;
    for (int i = tid; i < 16 * 32; i += blockDim.x) sW[i] = W_node[i];
    if (tid < 64) { sSh[tid] = shape_emb[tid]; sCo[tid] = color_emb[tid]; }
    if (tid < 32) sB[tid] = b_node[tid];
    __syncthreads();

    int node = blockIdx.x * blockDim.x + tid;
    if (node >= N_NODES) return;

    int si = x[node * 2 + 0];
    int ci = x[node * 2 + 1];
    float in[16];
#pragma unroll
    for (int j = 0; j < 8; ++j) { in[j] = sSh[si * 8 + j]; in[8 + j] = sCo[ci * 8 + j]; }

    __bf16 res[32];
#pragma unroll
    for (int n = 0; n < 32; ++n) {
        float acc = sB[n];
#pragma unroll
        for (int k = 0; k < 16; ++k) acc = fmaf(in[k], sW[k * 32 + n], acc);
        res[n] = (__bf16)(acc > 0.0f ? acc : 0.0f);
    }
    v4bf* out = (v4bf*)&h0[(size_t)node * 32];
#pragma unroll
    for (int n = 0; n < 8; ++n) out[n] = *(v4bf*)&res[n * 4];
}

// ---------------------------------------------------------------------------
// 2) edge scatter-add: agg[dst] += h[src]; bf16 gather (halves HBM gather
//    bytes), f32 atomics into L2-resident agg
// ---------------------------------------------------------------------------
template <int K>
__global__ void scatter_kernel(const __bf16* __restrict__ h,
                               const int*    __restrict__ edge_index, // (2,E)
                               float*        __restrict__ agg)
{
    constexpr int GROUPS = K / 4;
    constexpr int GSH    = (K == 32) ? 3 : 4;
    long long t = (long long)blockIdx.x * blockDim.x + threadIdx.x;
    long long total = (long long)N_EDGES * GROUPS;
    if (t >= total) return;
    int e = (int)(t >> GSH);
    int g = (int)(t & (GROUPS - 1)) * 4;
    int s = edge_index[e];
    int d = edge_index[N_EDGES + e];
    const v4bf v = *(const v4bf*)&h[(size_t)s * K + g];
    float* p = &agg[(size_t)d * K + g];
    atomicAdd(p + 0, (float)v[0]);
    atomicAdd(p + 1, (float)v[1]);
    atomicAdd(p + 2, (float)v[2]);
    atomicAdd(p + 3, (float)v[3]);
}

// ---------------------------------------------------------------------------
// 3) graph-conv GEMM: out = relu(agg @ W_rel + h @ W_root + b), bf16 WMMA.
//    Per block: 32 nodes x 64 cols; 8 waves = 2(M) x 4(N) tiles of 16x16.
//    h tile arrives via TDM (wave 0 issues descriptor, s_wait_tensorcnt);
//    agg tile converted f32->bf16 with packed stores; weights transposed so
//    A and B fragments are two contiguous 16B LDS loads per lane (ISA 7.12.2).
// ---------------------------------------------------------------------------
template <int K>
__global__ void conv_wmma_kernel(const float*  __restrict__ agg,     // (N,K) f32
                                 const __bf16* __restrict__ h,       // (N,K) bf16
                                 const float*  __restrict__ W_rel,   // (K,64)
                                 const float*  __restrict__ W_root,  // (K,64)
                                 const float*  __restrict__ bias,    // (64,)
                                 __bf16*       __restrict__ out)     // (N,64) bf16
{
    __shared__ __attribute__((aligned(16))) __bf16 sWrel [64 * K];  // [n][k]
    __shared__ __attribute__((aligned(16))) __bf16 sWroot[64 * K];  // [n][k]
    __shared__ __attribute__((aligned(16))) __bf16 sAgg  [32 * K];  // [r][k]
    __shared__ __attribute__((aligned(16))) __bf16 sH    [32 * K];  // [r][k]

    const int tid  = threadIdx.x;
    const int base = blockIdx.x * 32;

    // kick off async TDM copy of the h tile (overlaps manual staging below)
    if (tid < 32) {
        tdm_load_tile_bf16<K>(&h[(size_t)base * K],
                              (unsigned)(uintptr_t)&sH[0],
                              (unsigned)(N_NODES - base));
    }

    // stage transposed weights (K,64) -> [n*K + k]
    for (int idx = tid; idx < K * 64; idx += 256) {
        int k = idx >> 6, n = idx & 63;
        sWrel [n * K + k] = (__bf16)W_rel [idx];
        sWroot[n * K + k] = (__bf16)W_root[idx];
    }
    // stage agg tile: float4 load -> packed 4x bf16 -> 8B LDS store
    for (int idx4 = tid; idx4 < (32 * K) / 4; idx4 += 256) {
        int idx = idx4 * 4;
        int r = idx / K, k = idx - r * K;
        int node = base + r;
        float4 v = (node < N_NODES) ? *(const float4*)&agg[(size_t)node * K + k]
                                    : make_float4(0.f, 0.f, 0.f, 0.f);
        union { __bf16 b[4]; uint2 u; } t;
        t.b[0] = (__bf16)v.x; t.b[1] = (__bf16)v.y;
        t.b[2] = (__bf16)v.z; t.b[3] = (__bf16)v.w;
        *(uint2*)&sAgg[idx] = t.u;
    }
    if (tid < 32) __builtin_amdgcn_s_wait_tensorcnt(0);
    __syncthreads();

    const int wave = tid >> 5;
    const int lane = tid & 31;
    const int mt = wave >> 2;              // 0..1 : M tile
    const int nt = wave & 3;               // 0..3 : N tile
    const int l15  = lane & 15;
    const int hi16 = lane >> 4;
    const int kb   = hi16 * 8;             // K sub-offset per ISA A/B layout
    const int arow = mt * 16 + l15;
    const int bcol = nt * 16 + l15;

    v8f c = {};
#pragma unroll
    for (int kc = 0; kc < K; kc += 32) {
        FragBF16 a, b;
        a.h8[0] = *(const v8bf*)&sAgg [arow * K + kc + kb];
        a.h8[1] = *(const v8bf*)&sAgg [arow * K + kc + kb + 16];
        b.h8[0] = *(const v8bf*)&sWrel[bcol * K + kc + kb];
        b.h8[1] = *(const v8bf*)&sWrel[bcol * K + kc + kb + 16];
        c = __builtin_amdgcn_wmma_f32_16x16x32_bf16(false, a.v, false, b.v,
                                                    (short)0, c, false, false);
        a.h8[0] = *(const v8bf*)&sH    [arow * K + kc + kb];
        a.h8[1] = *(const v8bf*)&sH    [arow * K + kc + kb + 16];
        b.h8[0] = *(const v8bf*)&sWroot[bcol * K + kc + kb];
        b.h8[1] = *(const v8bf*)&sWroot[bcol * K + kc + kb + 16];
        c = __builtin_amdgcn_wmma_f32_16x16x32_bf16(false, a.v, false, b.v,
                                                    (short)0, c, false, false);
    }

    // epilogue: D layout -> N = lane&15 (+nt*16), M = r + 8*hi16 (+mt*16)
    const int ncol = nt * 16 + l15;
    const float bv = bias[ncol];
    const int m0 = base + mt * 16 + hi16 * 8;
#pragma unroll
    for (int r = 0; r < 8; ++r) {
        int m = m0 + r;
        if (m < N_NODES) {
            float v = c[r] + bv;
            out[(size_t)m * 64 + ncol] = (__bf16)(v > 0.0f ? v : 0.0f);
        }
    }
}

// ---------------------------------------------------------------------------
// 4) mean-pool: atomic segment sums over batch + counts (bf16 reads)
// ---------------------------------------------------------------------------
__global__ void pool_kernel(const __bf16* __restrict__ h2,     // (N,64) bf16
                            const int*    __restrict__ batch,  // (N,)
                            float*        __restrict__ pooled, // (G,64)
                            float*        __restrict__ counts) // (G,)
{
    long long t = (long long)blockIdx.x * blockDim.x + threadIdx.x;
    long long total = (long long)N_NODES * 16;
    if (t >= total) return;
    int node = (int)(t >> 4);
    int g4   = (int)(t & 15) * 4;
    int b = batch[node];
    const v4bf v = *(const v4bf*)&h2[(size_t)node * 64 + g4];
    float* p = &pooled[(size_t)b * 64 + g4];
    atomicAdd(p + 0, (float)v[0]);
    atomicAdd(p + 1, (float)v[1]);
    atomicAdd(p + 2, (float)v[2]);
    atomicAdd(p + 3, (float)v[3]);
    if (g4 == 0) atomicAdd(&counts[b], 1.0f);
}

// ---------------------------------------------------------------------------
// 5) classifier: out = (pooled / max(counts,1)) @ W_cls + b_cls  (f32 out)
// ---------------------------------------------------------------------------
__global__ void cls_kernel(const float* __restrict__ pooled,
                           const float* __restrict__ counts,
                           const float* __restrict__ W_cls,   // (64,10)
                           const float* __restrict__ b_cls,   // (10,)
                           float*       __restrict__ out)     // (G,10)
{
    int t = blockIdx.x * blockDim.x + threadIdx.x;
    if (t >= N_GRAPHS * NUM_CLASSES) return;
    int g   = t / NUM_CLASSES;
    int cls = t - g * NUM_CLASSES;
    float cnt = counts[g];
    float inv = 1.0f / (cnt < 1.0f ? 1.0f : cnt);
    float acc = b_cls[cls];
    const float* pr = &pooled[(size_t)g * 64];
#pragma unroll
    for (int k = 0; k < 64; ++k)
        acc = fmaf(pr[k] * inv, W_cls[k * NUM_CLASSES + cls], acc);
    out[t] = acc;
}

// ---------------------------------------------------------------------------
// launch
// ---------------------------------------------------------------------------
extern "C" void kernel_launch(void* const* d_in, const int* in_sizes, int n_in,
                              void* d_out, int out_size, void* d_ws, size_t ws_size,
                              hipStream_t stream) {
    const int*   x         = (const int*)  d_in[0];
    const int*   edge_idx  = (const int*)  d_in[1];
    const int*   batch     = (const int*)  d_in[2];
    const float* shape_emb = (const float*)d_in[3];
    const float* color_emb = (const float*)d_in[4];
    const float* W_node    = (const float*)d_in[5];
    const float* b_node    = (const float*)d_in[6];
    const float* W1_rel    = (const float*)d_in[7];
    const float* W1_root   = (const float*)d_in[8];
    const float* b1        = (const float*)d_in[9];
    const float* W2_rel    = (const float*)d_in[10];
    const float* W2_root   = (const float*)d_in[11];
    const float* b2        = (const float*)d_in[12];
    const float* W_cls     = (const float*)d_in[13];
    const float* b_cls     = (const float*)d_in[14];
    float* out = (float*)d_out;

    // workspace layout (all offsets 256B-aligned)
    char* ws = (char*)d_ws;
    const size_t AGG_BYTES = (size_t)N_NODES * 64 * sizeof(float);   // 25.6 MB
    const size_t H0_BYTES  = (size_t)N_NODES * 32 * sizeof(__bf16);  //  6.4 MB
    const size_t H_BYTES   = (size_t)N_NODES * 64 * sizeof(__bf16);  // 12.8 MB
    float*  aggf   = (float*)ws;                                  // agg1/agg2
    __bf16* h0     = (__bf16*)(ws + AGG_BYTES);
    __bf16* h1     = (__bf16*)(ws + AGG_BYTES + H0_BYTES);
    __bf16* h2     = (__bf16*)(ws + AGG_BYTES + H0_BYTES + H_BYTES);
    float*  pooled = (float*)(ws + AGG_BYTES + H0_BYTES + 2 * H_BYTES);
    float*  counts = pooled + (size_t)N_GRAPHS * 64;

    const int blk = 256;
    auto cdiv = [](long long a, long long b) { return (int)((a + b - 1) / b); };

    // 1) embedding + node MLP
    embed_mlp_kernel<<<cdiv(N_NODES, blk), blk, 0, stream>>>(
        x, shape_emb, color_emb, W_node, b_node, h0);

    // 2) conv1
    fill_zero_kernel<<<cdiv((long long)N_NODES * 32, blk), blk, 0, stream>>>(
        aggf, (long long)N_NODES * 32);
    scatter_kernel<32><<<cdiv((long long)N_EDGES * 8, blk), blk, 0, stream>>>(
        h0, edge_idx, aggf);
    conv_wmma_kernel<32><<<cdiv(N_NODES, 32), 256, 0, stream>>>(
        aggf, h0, W1_rel, W1_root, b1, h1);

    // 3) conv2
    fill_zero_kernel<<<cdiv((long long)N_NODES * 64, blk), blk, 0, stream>>>(
        aggf, (long long)N_NODES * 64);
    scatter_kernel<64><<<cdiv((long long)N_EDGES * 16, blk), blk, 0, stream>>>(
        h1, edge_idx, aggf);
    conv_wmma_kernel<64><<<cdiv(N_NODES, 32), 256, 0, stream>>>(
        aggf, h1, W2_rel, W2_root, b2, h2);

    // 4) mean pool
    fill_zero_kernel<<<cdiv((long long)N_GRAPHS * 65, blk), blk, 0, stream>>>(
        pooled, (long long)N_GRAPHS * 65);   // pooled + counts contiguous
    pool_kernel<<<cdiv((long long)N_NODES * 16, blk), blk, 0, stream>>>(
        h2, batch, pooled, counts);

    // 5) classifier
    cls_kernel<<<cdiv(N_GRAPHS * NUM_CLASSES, blk), blk, 0, stream>>>(
        pooled, counts, W_cls, b_cls, out);
}
